// RSGNNv2_49409303773439
// MI455X (gfx1250) — matrix operand order repaired and checked by reading, not compiled, not run on hardware
//
#include <hip/hip_runtime.h>
#include <hip/hip_bf16.h>
#include <math.h>

// Problem constants (match reference)
#define B_   4
#define T_   256
#define N_   8
#define D_   128
#define NH_  4
#define ROWS (B_*T_*N_)          // 8192 tokens
#define ATT_SCALE 0.17677669529663687f   // 1/sqrt(32)

typedef __attribute__((ext_vector_type(16))) _Float16 v16h;
typedef __attribute__((ext_vector_type(8)))  float    v8f;

__device__ __forceinline__ float gelu_f(float x) {
    return 0.5f * x * (1.0f + erff(x * 0.7071067811865476f));
}

__device__ __forceinline__ v8f wmma16(v16h a, v16h b, v8f c) {
    // D(16x16,f32) = A(16x32,f16) * B(32x16,f16) + C
    return __builtin_amdgcn_wmma_f32_16x16x32_f16(false, a, false, b, (short)0, c, false, false);
}

// ---------------------------------------------------------------------------
// DPP16 row reductions (16-lane rows = the two EXEC halves of a wave32).
// quad_perm folds quads; row_ror:4/8 folds the quad-uniform values. Pure VALU.
// ---------------------------------------------------------------------------
__device__ __forceinline__ float dpp_sum16(float v) {
    v += __int_as_float(__builtin_amdgcn_update_dpp(__float_as_int(v), __float_as_int(v), 0xB1,  0xF, 0xF, true));
    v += __int_as_float(__builtin_amdgcn_update_dpp(__float_as_int(v), __float_as_int(v), 0x4E,  0xF, 0xF, true));
    v += __int_as_float(__builtin_amdgcn_update_dpp(__float_as_int(v), __float_as_int(v), 0x124, 0xF, 0xF, true));
    v += __int_as_float(__builtin_amdgcn_update_dpp(__float_as_int(v), __float_as_int(v), 0x128, 0xF, 0xF, true));
    return v;
}
__device__ __forceinline__ float dpp_max16(float v) {
    v = fmaxf(v, __int_as_float(__builtin_amdgcn_update_dpp(__float_as_int(v), __float_as_int(v), 0xB1,  0xF, 0xF, true)));
    v = fmaxf(v, __int_as_float(__builtin_amdgcn_update_dpp(__float_as_int(v), __float_as_int(v), 0x4E,  0xF, 0xF, true)));
    v = fmaxf(v, __int_as_float(__builtin_amdgcn_update_dpp(__float_as_int(v), __float_as_int(v), 0x124, 0xF, 0xF, true)));
    v = fmaxf(v, __int_as_float(__builtin_amdgcn_update_dpp(__float_as_int(v), __float_as_int(v), 0x128, 0xF, 0xF, true)));
    return v;
}

// ---------------------------------------------------------------------------
// Async copy helpers (ASYNCcnt-tracked LDS DMA, CDNA5 §15.18.3 op 98)
// ---------------------------------------------------------------------------
__device__ __forceinline__ void async_b128(unsigned lds_byte_off, const void* gaddr) {
    asm volatile("global_load_async_to_lds_b128 %0, %1, off"
                 :: "v"(lds_byte_off), "v"((unsigned long long)(uintptr_t)gaddr)
                 : "memory");
}

// ---------------------------------------------------------------------------
// Stage 1a: b1[b,t,m] = lin + 0.1*nl + bias  (tiny 8-wide model, t in [0,T-2])
// ---------------------------------------------------------------------------
__global__ __launch_bounds__(256) void b1_build(
    const float* __restrict__ vis, const float* __restrict__ r5,
    const float* __restrict__ A55, const float* __restrict__ blb,
    const float* __restrict__ W1, const float* __restrict__ b1v,
    const float* __restrict__ W2, const float* __restrict__ b2v,
    float* __restrict__ b1out)
{
    int idx = blockIdx.x * blockDim.x + threadIdx.x;
    const int total = B_ * (T_ - 1) * N_;
    if (idx >= total) return;
    int m = idx & 7;
    int t = (idx >> 3) % (T_ - 1);
    int b = idx / ((T_ - 1) * N_);
    float xv[N_];
#pragma unroll
    for (int n = 0; n < N_; ++n) xv[n] = vis[((b * T_) + t) * N_ + n];
    float lin = r5[m];
#pragma unroll
    for (int n = 0; n < N_; ++n) lin += xv[n] * A55[m * N_ + n];
    float nl = b2v[m];
    for (int j = 0; j < 32; ++j) {
        float h = b1v[j];
#pragma unroll
        for (int n = 0; n < N_; ++n) h += xv[n] * W1[j * N_ + n];
        nl += gelu_f(h) * W2[m * 32 + j];
    }
    b1out[((b * T_) + t) * N_ + m] = lin + 0.1f * nl + blb[m];
}

// ---------------------------------------------------------------------------
// Stage 1b: residuals R[b,t,n,scale] (scales 1,2,3,5), zero-padded at tail
// ---------------------------------------------------------------------------
__global__ __launch_bounds__(256) void resid_build(
    const float* __restrict__ vis, const float* __restrict__ b1,
    float* __restrict__ Rout)
{
    int idx = blockIdx.x * blockDim.x + threadIdx.x;
    if (idx >= ROWS) return;
    int n = idx & 7, t = (idx >> 3) & 255, b = idx >> 11;
    const int ks[4] = {1, 2, 3, 5};
    float lx0 = __logf(fmaxf(vis[((b * T_) + t) * N_ + n], 1e-6f));
#pragma unroll
    for (int s = 0; s < 4; ++s) {
        int k = ks[s];
        float res = 0.f;
        if (t < T_ - k) {
            float lx1 = __logf(fmaxf(vis[((b * T_) + t + k) * N_ + n], 1e-6f));
            float act = fminf(fmaxf(lx1 - lx0, -2.5f * (float)k), 2.0f * (float)k);
            float bk = 0.f;
            for (int i = 0; i < k; ++i) bk += b1[((b * T_) + t + i) * N_ + n];
            res = act - bk;
        }
        Rout[(size_t)idx * 4 + s] = res;
    }
}

// ---------------------------------------------------------------------------
// Stage 1c: 16-channel features (residuals + lags 1,2,3)
// ---------------------------------------------------------------------------
__global__ __launch_bounds__(256) void feats_build(
    const float* __restrict__ Rin, float* __restrict__ F)
{
    int idx = blockIdx.x * blockDim.x + threadIdx.x;
    if (idx >= ROWS) return;
    int n = idx & 7, t = (idx >> 3) & 255, b = idx >> 11;
#pragma unroll
    for (int g = 0; g < 4; ++g) {          // lag = g (0,1,2,3)
#pragma unroll
        for (int s = 0; s < 4; ++s) {
            float v = 0.f;
            if (t >= g) v = Rin[((((b * T_) + (t - g)) * N_) + n) * 4 + s];
            F[(size_t)idx * 16 + g * 4 + s] = v;
        }
    }
}

// ---------------------------------------------------------------------------
// GEMM A-tile loaders. f32: VGPR path with convert + K-guard (K=16 case).
// f16: two async b128 per thread (K is always a multiple of 32 here), fenced
// by async_fence() before the barrier.
// ---------------------------------------------------------------------------
__device__ __forceinline__ void gemm_load_a(
    const float* A, _Float16* As, int ar, int ac, int m0, int k0, int K) {
    const size_t rb = (size_t)(m0 + ar) * K;
#pragma unroll
    for (int i = 0; i < 16; ++i) {
        int k = k0 + ac + i;
        As[ar * 40 + ac + i] = (k < K) ? (_Float16)A[rb + k] : (_Float16)0.f;
    }
    if (k0 + 32 < K) __builtin_prefetch(&A[rb + k0 + 32 + ac], 0, 1);
}
__device__ __forceinline__ void gemm_load_a(
    const _Float16* A, _Float16* As, int ar, int ac, int m0, int k0, int K) {
    const _Float16* g = A + (size_t)(m0 + ar) * K + k0 + ac;
    unsigned off = (unsigned)((ar * 40 + ac) * 2);
    async_b128(off, g);
    async_b128(off + 16u, g + 8);
}
__device__ __forceinline__ void async_fence(const float*) {}
__device__ __forceinline__ void async_fence(const _Float16*) {
    asm volatile("s_wait_asynccnt 0x0" ::: "memory");
}

// ---------------------------------------------------------------------------
// Generic WMMA GEMM: out[m,n] = act( sum_k A[m,k]*W[n,k] + bias[n] )
// A: [M,K] (f32 or f16), W: [N,K] f32.
// Block 256 thr = 8 waves. Block tile 128x64: wave w = 16-row stripe times all
// four 16-col tiles -> 4 WMMAs per A fragment, 4x W-tile reuse from LDS.
// Dynamic LDS (offset 0): As [128][40] f16 @0, Ws [64][40] f16 @10240B.
// ---------------------------------------------------------------------------
#define GEMM_LDS_BYTES 15360
template <typename TA>
__global__ __launch_bounds__(256) void gemm_ws(
    const TA* __restrict__ A, const float* __restrict__ W,
    const float* __restrict__ bias, float* __restrict__ outF,
    _Float16* __restrict__ outH, int M, int N, int K, int act)
{
    extern __shared__ _Float16 gsm[];
    _Float16* As = gsm;            // [128][40]
    _Float16* Ws = gsm + 5120;     // [64][40]

    const int tid  = threadIdx.x;
    const int lane = tid & 31, wave = tid >> 5;   // wave = 16-row stripe id
    const int half = lane >> 4, l16 = lane & 15;
    const int m0 = blockIdx.x * 128, n0 = blockIdx.y * 64;

    v8f acc[4];
#pragma unroll
    for (int nt = 0; nt < 4; ++nt)
#pragma unroll
        for (int r = 0; r < 8; ++r) acc[nt][r] = 0.f;

    const int ar = tid >> 1, ac = (tid & 1) << 4;   // A: 128 rows x 32 k, 16/thr
    const int wr = tid >> 2, wc = (tid & 3) << 3;   // W: 64 rows x 32 k, 8/thr

    for (int k0 = 0; k0 < K; k0 += 32) {
        __syncthreads();
        gemm_load_a(A, As, ar, ac, m0, k0, K);      // f16 => async LDS DMA
#pragma unroll
        for (int i = 0; i < 8; ++i) {               // overlaps with async copy
            int k = k0 + wc + i;
            Ws[wr * 40 + wc + i] = (k < K) ? (_Float16)W[(size_t)(n0 + wr) * K + k]
                                           : (_Float16)0.f;
        }
        async_fence(A);
        __syncthreads();

        v16h af;
        // A fragment: lane(m=l16) holds K = 16*half + e, 2/VGPR
#pragma unroll
        for (int e = 0; e < 16; ++e) {
            int i = e >> 1, j = e & 1;
            int k = (half << 4) + e;
            af[e] = As[((wave << 4) + l16) * 40 + k];
        }
#pragma unroll
        for (int nt = 0; nt < 4; ++nt) {
            v16h bf;   // B fragment: lane(n=l16), K = 16*half + e
#pragma unroll
            for (int e = 0; e < 16; ++e)
                bf[e] = Ws[((nt << 4) + l16) * 40 + (half << 4) + e];
            acc[nt] = wmma16(af, bf, acc[nt]);
        }
    }

#pragma unroll
    for (int nt = 0; nt < 4; ++nt) {
        const int col = n0 + (nt << 4) + l16;
        const float bv = bias ? bias[col] : 0.f;
#pragma unroll
        for (int r = 0; r < 8; ++r) {
            int row = m0 + (wave << 4) + r + (half << 3);  // C/D: M = r + 8*half
            float v = acc[nt][r] + bv;
            if (act == 1) v = gelu_f(v);
            if (outF) outF[(size_t)row * N + col] = v;
            if (outH) outH[(size_t)row * N + col] = (_Float16)v;
        }
    }
}

// ---------------------------------------------------------------------------
// Flash attention (hd = 32). One wave = 16 query rows; key blocks of 32.
// Block = 128 thr (4 waves, same seq & head). K/V tiles double-buffered via
// GLOBAL_LOAD_ASYNC_TO_LDS_B128: block i+1 is in flight while block i computes
// (async loads retire in order per wave => s_wait_asynccnt 0x2). Softmax
// stats via DPP16. Dynamic LDS (offset 0):
//   stage p in {0,1}: Ks @ p*5120, Vs @ p*5120+2560 (bytes)
//   Ps @ 10240 bytes = [4][16][40] f16
// ---------------------------------------------------------------------------
#define FLASH_LDS_BYTES 15360
__global__ __launch_bounds__(128) void flash_attn(
    const _Float16* __restrict__ Q, const _Float16* __restrict__ K,
    const _Float16* __restrict__ V, _Float16* __restrict__ O,
    int Lq, int Lk, int ldq, int ldk, int ldv, int ldo,
    int nheads, float scale)
{
    extern __shared__ _Float16 fsm[];
    _Float16* Ps = fsm + 5120;           // [wave][row][40]

    const int tid  = threadIdx.x;
    const int lane = tid & 31, wave = tid >> 5;
    const int half = lane >> 4, l16 = lane & 15;

    const int qtiles = Lq >> 6;                 // 4 waves * 16 rows
    int x = blockIdx.x;
    const int qb = x % qtiles; x /= qtiles;
    const int h  = x % nheads; x /= nheads;
    const int s  = x;

    const size_t qbase = (size_t)s * Lq * ldq + h * 32;
    const size_t kbase = (size_t)s * Lk * ldk + h * 32;
    const size_t vbase = (size_t)s * Lk * ldv + h * 32;
    const size_t obase = (size_t)s * Lq * ldo + h * 32;

    const int qr0 = (qb << 6) + (wave << 4);

    // Q fragment (A layout, 16x32) straight from global
    v16h qf;
    const _Float16* qp = Q + qbase + (size_t)(qr0 + l16) * ldq;
#pragma unroll
    for (int e = 0; e < 16; ++e) {
        int i = e >> 1, j = e & 1;
        int k = (half << 4) + e;
        qf[e] = qp[k];
    }

    v8f o0, o1;
    float mi[8], li[8];
#pragma unroll
    for (int r = 0; r < 8; ++r) { o0[r] = 0.f; o1[r] = 0.f; mi[r] = -INFINITY; li[r] = 0.f; }

    // each of the 128 threads async-copies one 16B chunk of K and of V
    const int lr = tid >> 2, lc = (tid & 3) << 3;          // 32 keys x 32 hd
    const unsigned chunk = (unsigned)((lr * 40 + lc) * 2); // byte offset in tile

    // prologue: stage 0 in flight
    async_b128(chunk,         K + kbase + (size_t)lr * ldk + lc);
    async_b128(2560u + chunk, V + vbase + (size_t)lr * ldv + lc);

    for (int kb = 0; kb < Lk; kb += 32) {
        const int p = (kb >> 5) & 1;
        __syncthreads();   // buffer p^1 fully consumed by all waves
        if (kb + 32 < Lk) {
            unsigned nb = (unsigned)((p ^ 1) * 5120);
            async_b128(nb + chunk,         K + kbase + (size_t)(kb + 32 + lr) * ldk + lc);
            async_b128(nb + 2560u + chunk, V + vbase + (size_t)(kb + 32 + lr) * ldv + lc);
            asm volatile("s_wait_asynccnt 0x2" ::: "memory");  // stage p landed
        } else {
            asm volatile("s_wait_asynccnt 0x0" ::: "memory");
        }
        __syncthreads();   // all waves' stage-p tiles visible

        const _Float16* Ks = fsm + p * 2560;
        const _Float16* Vs = Ks + 1280;

        // S = Q * K^T : B(k=hd, n=key)
        v16h bk0, bk1;
#pragma unroll
        for (int e = 0; e < 16; ++e) {
            bk0[e] = Ks[l16 * 40 + (half << 4) + e];
            bk1[e] = Ks[(16 + l16) * 40 + (half << 4) + e];
        }
        v8f z;
#pragma unroll
        for (int r = 0; r < 8; ++r) z[r] = 0.f;
        v8f s0 = wmma16(qf, bk0, z);
        v8f s1 = wmma16(qf, bk1, z);

        // online softmax; row m = r + 8*half, cols spread over the 16 lanes
#pragma unroll
        for (int r = 0; r < 8; ++r) {
            float a0 = s0[r] * scale, a1 = s1[r] * scale;
            float mx = dpp_max16(fmaxf(a0, a1));
            float mn = fmaxf(mi[r], mx);
            float p0 = __expf(a0 - mn), p1 = __expf(a1 - mn);
            float sum = dpp_sum16(p0 + p1);
            float corr = __expf(mi[r] - mn);
            li[r] = li[r] * corr + sum;
            o0[r] *= corr; o1[r] *= corr;
            mi[r] = mn;
            Ps[(wave << 4 | (r + (half << 3))) * 40 + l16]      = (_Float16)p0;
            Ps[(wave << 4 | (r + (half << 3))) * 40 + 16 + l16] = (_Float16)p1;
        }

        // O += P * V : A = P (16x32 via per-wave LDS staging), B(k=key, n=hd)
        v16h pf, bv0, bv1;
#pragma unroll
        for (int e = 0; e < 16; ++e) {
            int i = e >> 1, j = e & 1;
            int k = (half << 4) + e;
            pf[e] = Ps[(wave << 4 | l16) * 40 + k];
        }
#pragma unroll
        for (int e = 0; e < 16; ++e) {
            bv0[e] = Vs[((half << 4) + e) * 40 + l16];
            bv1[e] = Vs[((half << 4) + e) * 40 + 16 + l16];
        }
        o0 = wmma16(pf, bv0, o0);
        o1 = wmma16(pf, bv1, o1);
    }

#pragma unroll
    for (int r = 0; r < 8; ++r) {
        float inv = 1.f / li[r];
        int row = qr0 + r + (half << 3);
        O[obase + (size_t)row * ldo + l16]      = (_Float16)(o0[r] * inv);
        O[obase + (size_t)row * ldo + 16 + l16] = (_Float16)(o1[r] * inv);
    }
}

// ---------------------------------------------------------------------------
// LayerNorm (+residual, +optional PE, +optional row permutation on store)
// order: 0 identity, 1 (b,n,t)->(b,t,n), 2 (b,t,n)->(b,n,t)
// Row reduction: DPP16 within 16-lane rows, tiny LDS combine across the 8 rows.
// ---------------------------------------------------------------------------
__global__ __launch_bounds__(128) void ln_pe(
    const float* __restrict__ X, const float* __restrict__ Rres,
    const float* __restrict__ g, const float* __restrict__ bb,
    float* __restrict__ outF, _Float16* __restrict__ outH,
    int order, int addPE)
{
    __shared__ float red[8];
    const int rin = blockIdx.x;
    const int d = threadIdx.x;
    const int lane = d & 31, wave = d >> 5;
    const int half = lane >> 4, l16 = lane & 15;

    float v = X[(size_t)rin * D_ + d] + (Rres ? Rres[(size_t)rin * D_ + d] : 0.f);

    float rs = dpp_sum16(v);
    if (l16 == 0) red[(wave << 1) | half] = rs;
    __syncthreads();
    float mean = (red[0] + red[1] + red[2] + red[3] +
                  red[4] + red[5] + red[6] + red[7]) * (1.f / D_);
    __syncthreads();

    float dv = v - mean;
    float rs2 = dpp_sum16(dv * dv);
    if (l16 == 0) red[(wave << 1) | half] = rs2;
    __syncthreads();
    float var = (red[0] + red[1] + red[2] + red[3] +
                 red[4] + red[5] + red[6] + red[7]) * (1.f / D_);

    float y = dv * rsqrtf(var + 1e-5f) * g[d] + bb[d];
    if (addPE) {                       // PE added after LN (input rows are (b,t,n))
        int t = (rin >> 3) & 255;
        int i = d >> 1;
        float freq = __expf(-(float)(2 * i) * (9.210340371976184f / (float)D_));
        y += (d & 1) ? __cosf((float)t * freq) : __sinf((float)t * freq);
    }
    int rout = rin;
    if (order == 1) {            // (b,n,t) -> (b,t,n)
        int t = rin & 255, n = (rin >> 8) & 7, b = rin >> 11;
        rout = ((b << 8) + t) * 8 + n;
    } else if (order == 2) {     // (b,t,n) -> (b,n,t)
        int n = rin & 7, t = (rin >> 3) & 255, b = rin >> 11;
        rout = ((b << 3) + n) * 256 + t;
    }
    if (outF) outF[(size_t)rout * D_ + d] = y;
    if (outH) outH[(size_t)rout * D_ + d] = (_Float16)y;
}

// ---------------------------------------------------------------------------
// Pack cat = [s1 | r] as f16 [ROWS, 256]
// ---------------------------------------------------------------------------
__global__ __launch_bounds__(256) void pack_cat(
    const _Float16* __restrict__ s1, const _Float16* __restrict__ r,
    _Float16* __restrict__ cat)
{
    int idx = blockIdx.x * blockDim.x + threadIdx.x;
    if (idx >= ROWS * D_) return;
    int row = idx >> 7, d = idx & 127;
    cat[(size_t)row * 256 + d]       = s1[idx];
    cat[(size_t)row * 256 + 128 + d] = r[idx];
}

// ---------------------------------------------------------------------------
// Host orchestration
// ---------------------------------------------------------------------------
extern "C" void kernel_launch(void* const* d_in, const int* in_sizes, int n_in,
                              void* d_out, int out_size, void* d_ws, size_t ws_size,
                              hipStream_t stream)
{
    const float* vis       = (const float*)d_in[0];
    const float* semb      = (const float*)d_in[1];
    const float* r5        = (const float*)d_in[2];
    const float* A55       = (const float*)d_in[3];
    const float* bl_bias   = (const float*)d_in[4];
    const float* nl_W1     = (const float*)d_in[5];
    const float* nl_b1     = (const float*)d_in[6];
    const float* nl_W2     = (const float*)d_in[7];
    const float* nl_b2     = (const float*)d_in[8];
    const float* mlp_W1    = (const float*)d_in[9];
    const float* mlp_b1    = (const float*)d_in[10];
    const float* mlp_W2    = (const float*)d_in[11];
    const float* mlp_b2    = (const float*)d_in[12];
    const float* mlp_ln_g  = (const float*)d_in[13];
    const float* mlp_ln_b  = (const float*)d_in[14];
    const float* eqkvW     = (const float*)d_in[15];
    const float* eqkvB     = (const float*)d_in[16];
    const float* eoutW     = (const float*)d_in[17];
    const float* eoutB     = (const float*)d_in[18];
    const float* eln1g     = (const float*)d_in[19];
    const float* eln1b     = (const float*)d_in[20];
    const float* effW1     = (const float*)d_in[21];
    const float* effB1     = (const float*)d_in[22];
    const float* effW2     = (const float*)d_in[23];
    const float* effB2     = (const float*)d_in[24];
    const float* eln2g     = (const float*)d_in[25];
    const float* eln2b     = (const float*)d_in[26];
    const float* caWq      = (const float*)d_in[27];
    const float* caBq      = (const float*)d_in[28];
    const float* caWk      = (const float*)d_in[29];
    const float* caBk      = (const float*)d_in[30];
    const float* caWv      = (const float*)d_in[31];
    const float* caBv      = (const float*)d_in[32];
    const float* caWo      = (const float*)d_in[33];
    const float* caBo      = (const float*)d_in[34];
    const float* caln1g    = (const float*)d_in[35];
    const float* caln1b    = (const float*)d_in[36];
    const float* caffW1    = (const float*)d_in[37];
    const float* caffB1    = (const float*)d_in[38];
    const float* caffW2    = (const float*)d_in[39];
    const float* caffB2    = (const float*)d_in[40];
    const float* caln2g    = (const float*)d_in[41];
    const float* caln2b    = (const float*)d_in[42];

    // bump allocator over workspace
    char* p = (char*)d_ws;
    auto alloc = [&](size_t bytes) -> void* {
        void* r = (void*)p;
        p += (bytes + 255) & ~(size_t)255;
        return r;
    };
    float*     b1buf = (float*)    alloc((size_t)B_ * T_ * N_ * 4);
    float*     Rbuf  = (float*)    alloc((size_t)ROWS * 4 * 4);
    float*     feats = (float*)    alloc((size_t)ROWS * 16 * 4);
    float*     tmp32 = (float*)    alloc((size_t)ROWS * D_ * 4);
    float*     xt32  = (float*)    alloc((size_t)ROWS * D_ * 4);
    _Float16*  xt16  = (_Float16*) alloc((size_t)ROWS * D_ * 2);
    _Float16*  qkv   = (_Float16*) alloc((size_t)ROWS * 384 * 2);
    _Float16*  obuf  = (_Float16*) alloc((size_t)ROWS * D_ * 2);
    float*     r32   = (float*)    alloc((size_t)ROWS * D_ * 4);
    _Float16*  r16   = (_Float16*) alloc((size_t)ROWS * D_ * 2);
    float*     s132  = (float*)    alloc((size_t)ROWS * D_ * 4);
    _Float16*  s116  = (_Float16*) alloc((size_t)ROWS * D_ * 2);
    _Float16*  cat   = (_Float16*) alloc((size_t)ROWS * 256 * 2);
    _Float16*  hbuf  = (_Float16*) alloc((size_t)ROWS * 512 * 2);
    (void)ws_size; (void)n_in; (void)in_sizes; (void)out_size;

    // ---- Stage 1: residual features ----
    b1_build<<<32, 256, 0, stream>>>(vis, r5, A55, bl_bias, nl_W1, nl_b1, nl_W2, nl_b2, b1buf);
    resid_build<<<32, 256, 0, stream>>>(vis, b1buf, Rbuf);
    feats_build<<<32, 256, 0, stream>>>(Rbuf, feats);

    // ---- Stage 2: token MLP (16->128->128) + LN + PE, transpose to (b,n,t) ----
    gemm_ws<float><<<dim3(ROWS/128, 2), 256, GEMM_LDS_BYTES, stream>>>(
        feats, mlp_W1, mlp_b1, nullptr, hbuf, ROWS, 128, 16, 1);
    gemm_ws<_Float16><<<dim3(ROWS/128, 2), 256, GEMM_LDS_BYTES, stream>>>(
        hbuf, mlp_W2, mlp_b2, tmp32, nullptr, ROWS, 128, 128, 1);
    ln_pe<<<ROWS, 128, 0, stream>>>(tmp32, nullptr, mlp_ln_g, mlp_ln_b, xt32, xt16, 2, 1);

    // ---- Stage 3: two encoder layers over 32 sequences of T=256 ----
    for (int l = 0; l < 2; ++l) {
        gemm_ws<_Float16><<<dim3(ROWS/128, 6), 256, GEMM_LDS_BYTES, stream>>>(
            xt16, eqkvW + (size_t)l * 384 * 128, eqkvB + (size_t)l * 384,
            nullptr, qkv, ROWS, 384, 128, 0);
        flash_attn<<<512, 128, FLASH_LDS_BYTES, stream>>>(
            qkv, qkv + 128, qkv + 256, obuf,
            256, 256, 384, 384, 384, 128, NH_, ATT_SCALE);
        gemm_ws<_Float16><<<dim3(ROWS/128, 2), 256, GEMM_LDS_BYTES, stream>>>(
            obuf, eoutW + (size_t)l * 128 * 128, eoutB + (size_t)l * 128,
            tmp32, nullptr, ROWS, 128, 128, 0);
        ln_pe<<<ROWS, 128, 0, stream>>>(xt32, tmp32, eln1g + l * 128, eln1b + l * 128,
                                        xt32, xt16, 0, 0);
        gemm_ws<_Float16><<<dim3(ROWS/128, 4), 256, GEMM_LDS_BYTES, stream>>>(
            xt16, effW1 + (size_t)l * 256 * 128, effB1 + (size_t)l * 256,
            nullptr, hbuf, ROWS, 256, 128, 1);
        gemm_ws<_Float16><<<dim3(ROWS/128, 2), 256, GEMM_LDS_BYTES, stream>>>(
            hbuf, effW2 + (size_t)l * 128 * 256, effB2 + (size_t)l * 128,
            tmp32, nullptr, ROWS, 128, 256, 0);
        if (l == 0) {
            ln_pe<<<ROWS, 128, 0, stream>>>(xt32, tmp32, eln2g, eln2b, xt32, xt16, 0, 0);
        } else {   // final layer: store r in (b,t,n) order for cross-attention
            ln_pe<<<ROWS, 128, 0, stream>>>(xt32, tmp32, eln2g + 128, eln2b + 128,
                                            r32, r16, 1, 0);
        }
    }

    // ---- Stage 4: cross attention (state_emb queries, residual_emb kv, L=2048) ----
    _Float16* qb = qkv;
    _Float16* kb = qkv + (size_t)ROWS * 128;
    _Float16* vb = qkv + (size_t)2 * ROWS * 128;
    gemm_ws<float><<<dim3(ROWS/128, 2), 256, GEMM_LDS_BYTES, stream>>>(
        semb, caWq, caBq, nullptr, qb, ROWS, 128, 128, 0);
    gemm_ws<_Float16><<<dim3(ROWS/128, 2), 256, GEMM_LDS_BYTES, stream>>>(
        r16, caWk, caBk, nullptr, kb, ROWS, 128, 128, 0);
    gemm_ws<_Float16><<<dim3(ROWS/128, 2), 256, GEMM_LDS_BYTES, stream>>>(
        r16, caWv, caBv, nullptr, vb, ROWS, 128, 128, 0);
    flash_attn<<<512, 128, FLASH_LDS_BYTES, stream>>>(
        qb, kb, vb, obuf, 2048, 2048, 128, 128, 128, 128, NH_, ATT_SCALE);
    gemm_ws<_Float16><<<dim3(ROWS/128, 2), 256, GEMM_LDS_BYTES, stream>>>(
        obuf, caWo, caBo, tmp32, nullptr, ROWS, 128, 128, 0);
    ln_pe<<<ROWS, 128, 0, stream>>>(semb, tmp32, caln1g, caln1b, s132, s116, 0, 0);

    // ---- Stage 5: concat-FF (256->512->128) + final LN -> d_out ----
    pack_cat<<<(ROWS * D_) / 256, 256, 0, stream>>>(s116, r16, cat);
    gemm_ws<_Float16><<<dim3(ROWS/128, 8), 256, GEMM_LDS_BYTES, stream>>>(
        cat, caffW1, caffB1, nullptr, hbuf, ROWS, 512, 256, 1);
    gemm_ws<_Float16><<<dim3(ROWS/128, 2), 256, GEMM_LDS_BYTES, stream>>>(
        hbuf, caffW2, caffB2, tmp32, nullptr, ROWS, 128, 512, 0);
    ln_pe<<<ROWS, 128, 0, stream>>>(s132, tmp32, caln2g, caln2b,
                                    (float*)d_out, nullptr, 0, 0);
}
